// Informer_75170517615190
// MI455X (gfx1250) — compile-verified, matrix operand order checked by reading, not run on hardware
//
#include <hip/hip_runtime.h>

// ---------------- types / constants ----------------
typedef __bf16 bf16_t;
typedef __attribute__((ext_vector_type(16))) __bf16 bf16x16;
typedef __attribute__((ext_vector_type(8)))  float  v8f;

#define DMODEL 512
#define NHEAD  8
#define DK     64
#define NDFF   2048
#define NB     16
#define TOPK_K 16
#define FLT_NEG (-3.0e38f)
#define NEG_BIG (-1.0e9f)

static __device__ __forceinline__ bf16_t f2bf(float f) {
    unsigned u = __builtin_bit_cast(unsigned, f);
    unsigned r = (u + 0x7FFFu + ((u >> 16) & 1u)) >> 16;
    return __builtin_bit_cast(bf16_t, (unsigned short)r);
}
static __device__ __forceinline__ bf16_t bf_zero() {
    return __builtin_bit_cast(bf16_t, (unsigned short)0);
}

// ---------------- WMMA fragment helpers ----------------
// v_wmma_f32_16x16x32_bf16 operand layout (A: [16 x 32], row per lane; B loaded
// identically from [N,K]): lane = row mod 16, lane>=16 adds +8 to K base;
// element j -> K = base + (j<8 ? j : j+8).  Loads are UNCONDITIONAL: all
// staging buffers live in workspace padded by 128KB, so tile overshoot reads
// stay in-bounds and only feed output rows/cols that are never stored.
static __device__ __forceinline__ bf16x16 ld_frag(const bf16_t* __restrict__ p) {
    bf16x16 f;
#pragma unroll
    for (int j = 0; j < 8; ++j) { f[j] = p[j]; f[j + 8] = p[j + 16]; }
    return f;
}
// Zero out elements whose local k (relative to this 32-chunk) >= rem.
static __device__ __forceinline__ bf16x16 mask_frag(bf16x16 f, int kbase, int rem) {
#pragma unroll
    for (int j = 0; j < 16; ++j) {
        int kloc = kbase + (j < 8 ? j : j + 8);
        if (kloc >= rem) f[j] = bf_zero();
    }
    return f;
}

static __device__ __forceinline__ void
store_sub(const v8f& c, float* __restrict__ C, bf16_t* __restrict__ Cb,
          const float* __restrict__ bias, int m0, int n0, int M, int N, int ldc,
          float alpha, int relu) {
    int lane = threadIdx.x & 31;
    int n = n0 + (lane & 15);
    if (n >= N) return;
    int mb = m0 + ((lane >> 4) << 3);
    float bv = bias ? bias[n] : 0.0f;
#pragma unroll
    for (int r = 0; r < 8; ++r) {
        int m = mb + r;
        if (m < M) {
            float v = c[r] * alpha + bv;
            if (relu) v = v > 0.0f ? v : 0.0f;
            C[(size_t)m * ldc + n] = v;
            if (Cb) Cb[(size_t)m * ldc + n] = f2bf(v);
        }
    }
}

// ---------------- batched GEMM: C = alpha*(A[M,K] x B[N,K]^T) + bias ----------------
// Block = 128 threads (4 waves, 2x2); block tile 64(M) x 128(N); wave tile 32x64:
// per K-step 6 fragment loads feed 8 WMMAs.
__global__ __launch_bounds__(128) void k_gemm(
    const bf16_t* __restrict__ A, const bf16_t* __restrict__ Bm,
    float* __restrict__ C, bf16_t* __restrict__ Cb, const float* __restrict__ bias,
    int M, int N, int K, int lda, int ldb, int ldc,
    long long sA, long long sB, long long sC, float alpha, int relu) {
    int z = blockIdx.z;
    const bf16_t* Az = A  + (size_t)z * sA;
    const bf16_t* Bz = Bm + (size_t)z * sB;
    float*  Cz  = C + (size_t)z * sC;
    bf16_t* Cbz = Cb ? Cb + (size_t)z * sC : (bf16_t*)0;

    int wave = threadIdx.x >> 5;
    int wm = wave >> 1, wn = wave & 1;
    int m0 = blockIdx.y * 64 + wm * 32;
    int n0 = blockIdx.x * 128 + wn * 64;
    if (m0 >= M || n0 >= N) return;   // wave-uniform: EXEC stays all-ones for WMMA

    int lane = threadIdx.x & 31;
    int kb = (lane >> 4) << 3;        // +8 K offset for upper half-wave
    const bf16_t* pa0 = Az + (size_t)(m0 + (lane & 15)) * lda + kb;
    const bf16_t* pa1 = pa0 + (size_t)16 * lda;
    const bf16_t* pb0 = Bz + (size_t)(n0 + (lane & 15)) * ldb + kb;
    const bf16_t* pb1 = pb0 + (size_t)16 * ldb;
    const bf16_t* pb2 = pb0 + (size_t)32 * ldb;
    const bf16_t* pb3 = pb0 + (size_t)48 * ldb;

    v8f c00 = {}, c01 = {}, c02 = {}, c03 = {};
    v8f c10 = {}, c11 = {}, c12 = {}, c13 = {};
    int Kmain = K & ~31;
    for (int k0 = 0; k0 < Kmain; k0 += 32) {
        __builtin_prefetch((const void*)(pa0 + 64), 0, 3);
        __builtin_prefetch((const void*)(pb0 + 64), 0, 3);
        bf16x16 a0 = ld_frag(pa0);
        bf16x16 a1 = ld_frag(pa1);
        bf16x16 b0 = ld_frag(pb0);
        bf16x16 b1 = ld_frag(pb1);
        bf16x16 b2 = ld_frag(pb2);
        bf16x16 b3 = ld_frag(pb3);
        c00 = __builtin_amdgcn_wmma_f32_16x16x32_bf16(false, a0, false, b0, (short)0, c00, false, false);
        c01 = __builtin_amdgcn_wmma_f32_16x16x32_bf16(false, a0, false, b1, (short)0, c01, false, false);
        c02 = __builtin_amdgcn_wmma_f32_16x16x32_bf16(false, a0, false, b2, (short)0, c02, false, false);
        c03 = __builtin_amdgcn_wmma_f32_16x16x32_bf16(false, a0, false, b3, (short)0, c03, false, false);
        c10 = __builtin_amdgcn_wmma_f32_16x16x32_bf16(false, a1, false, b0, (short)0, c10, false, false);
        c11 = __builtin_amdgcn_wmma_f32_16x16x32_bf16(false, a1, false, b1, (short)0, c11, false, false);
        c12 = __builtin_amdgcn_wmma_f32_16x16x32_bf16(false, a1, false, b2, (short)0, c12, false, false);
        c13 = __builtin_amdgcn_wmma_f32_16x16x32_bf16(false, a1, false, b3, (short)0, c13, false, false);
        pa0 += 32; pa1 += 32; pb0 += 32; pb1 += 32; pb2 += 32; pb3 += 32;
    }
    if (K & 31) {                      // wave-uniform K-remainder with value masking
        int rem = K - Kmain;
        bf16x16 a0 = mask_frag(ld_frag(pa0), kb, rem);
        bf16x16 a1 = mask_frag(ld_frag(pa1), kb, rem);
        bf16x16 b0 = mask_frag(ld_frag(pb0), kb, rem);
        bf16x16 b1 = mask_frag(ld_frag(pb1), kb, rem);
        bf16x16 b2 = mask_frag(ld_frag(pb2), kb, rem);
        bf16x16 b3 = mask_frag(ld_frag(pb3), kb, rem);
        c00 = __builtin_amdgcn_wmma_f32_16x16x32_bf16(false, a0, false, b0, (short)0, c00, false, false);
        c01 = __builtin_amdgcn_wmma_f32_16x16x32_bf16(false, a0, false, b1, (short)0, c01, false, false);
        c02 = __builtin_amdgcn_wmma_f32_16x16x32_bf16(false, a0, false, b2, (short)0, c02, false, false);
        c03 = __builtin_amdgcn_wmma_f32_16x16x32_bf16(false, a0, false, b3, (short)0, c03, false, false);
        c10 = __builtin_amdgcn_wmma_f32_16x16x32_bf16(false, a1, false, b0, (short)0, c10, false, false);
        c11 = __builtin_amdgcn_wmma_f32_16x16x32_bf16(false, a1, false, b1, (short)0, c11, false, false);
        c12 = __builtin_amdgcn_wmma_f32_16x16x32_bf16(false, a1, false, b2, (short)0, c12, false, false);
        c13 = __builtin_amdgcn_wmma_f32_16x16x32_bf16(false, a1, false, b3, (short)0, c13, false, false);
    }
    store_sub(c00, Cz, Cbz, bias, m0,      n0,      M, N, ldc, alpha, relu);
    store_sub(c01, Cz, Cbz, bias, m0,      n0 + 16, M, N, ldc, alpha, relu);
    store_sub(c02, Cz, Cbz, bias, m0,      n0 + 32, M, N, ldc, alpha, relu);
    store_sub(c03, Cz, Cbz, bias, m0,      n0 + 48, M, N, ldc, alpha, relu);
    store_sub(c10, Cz, Cbz, bias, m0 + 16, n0,      M, N, ldc, alpha, relu);
    store_sub(c11, Cz, Cbz, bias, m0 + 16, n0 + 16, M, N, ldc, alpha, relu);
    store_sub(c12, Cz, Cbz, bias, m0 + 16, n0 + 32, M, N, ldc, alpha, relu);
    store_sub(c13, Cz, Cbz, bias, m0 + 16, n0 + 48, M, N, ldc, alpha, relu);
}

// ---------------- top-k / mask + softmax (one wave32 per row) ----------------
__global__ __launch_bounds__(256) void k_softmax(
    float* __restrict__ S, bf16_t* __restrict__ Sb,
    int Lq, int Lk, int mode /*0 plain,1 topk,2 causal*/, int topk, int rowsTotal) {
    int row = blockIdx.x * 8 + (threadIdx.x >> 5);
    if (row >= rowsTotal) return;
    int lane = threadIdx.x & 31;
    int q = row % Lq;
    const float* rp = S + (size_t)row * Lk;
    float v[16];
#pragma unroll
    for (int j = 0; j < 16; ++j) {
        int col = lane + 32 * j;
        float x = FLT_NEG;
        if (col < Lk) {
            x = rp[col];
            if (mode == 2 && col > q) x += NEG_BIG;
        }
        v[j] = x;
    }
    if (mode == 1) {
        float t[16];
#pragma unroll
        for (int j = 0; j < 16; ++j) t[j] = v[j];
        float kth = FLT_NEG;
        for (int it = 0; it < topk; ++it) {
            float lm = FLT_NEG; int li = 0;
#pragma unroll
            for (int j = 0; j < 16; ++j) if (t[j] > lm) { lm = t[j]; li = j; }
            float bm = lm; int bl = lane, bi = li;
            for (int o = 16; o; o >>= 1) {
                float om = __shfl_xor(bm, o, 32);
                int   ol = __shfl_xor(bl, o, 32);
                int   oi = __shfl_xor(bi, o, 32);
                if (om > bm || (om == bm && ol < bl)) { bm = om; bl = ol; bi = oi; }
            }
            kth = bm;
            if (lane == bl) t[bi] = FLT_NEG;
        }
#pragma unroll
        for (int j = 0; j < 16; ++j) if (v[j] < kth) v[j] = NEG_BIG;
    }
    float mx = FLT_NEG;
#pragma unroll
    for (int j = 0; j < 16; ++j) mx = fmaxf(mx, v[j]);
    for (int o = 16; o; o >>= 1) mx = fmaxf(mx, __shfl_xor(mx, o, 32));
    float e[16]; float sum = 0.0f;
#pragma unroll
    for (int j = 0; j < 16; ++j) { e[j] = __expf(v[j] - mx); sum += e[j]; }
    for (int o = 16; o; o >>= 1) sum += __shfl_xor(sum, o, 32);
    float inv = 1.0f / sum;
#pragma unroll
    for (int j = 0; j < 16; ++j) {
        int col = lane + 32 * j;
        if (col < Lk) Sb[(size_t)row * Lk + col] = f2bf(e[j] * inv);
    }
}

// ---------------- add + layernorm (row per block), dual f32/bf16 output ----------------
__global__ __launch_bounds__(128) void k_add_ln(
    const float* __restrict__ X, const float* __restrict__ Dl, const float* __restrict__ lp,
    float* __restrict__ Yf, bf16_t* __restrict__ Yb, int M) {
    int row = blockIdx.x;
    if (row >= M) return;
    __shared__ float sh[8];
    const float* xr = X + (size_t)row * DMODEL;
    const float* dr = Dl ? Dl + (size_t)row * DMODEL : (const float*)0;
    float loc[4]; float s = 0.0f, s2 = 0.0f;
#pragma unroll
    for (int i = 0; i < 4; ++i) {
        int c = threadIdx.x + 128 * i;
        float v = xr[c] + (dr ? dr[c] : 0.0f);
        loc[i] = v; s += v; s2 += v * v;
    }
    for (int o = 16; o; o >>= 1) { s += __shfl_xor(s, o, 32); s2 += __shfl_xor(s2, o, 32); }
    int wid = threadIdx.x >> 5;
    if ((threadIdx.x & 31) == 0) { sh[wid] = s; sh[4 + wid] = s2; }
    __syncthreads();
    s  = sh[0] + sh[1] + sh[2] + sh[3];
    s2 = sh[4] + sh[5] + sh[6] + sh[7];
    float mean = s * (1.0f / DMODEL);
    float var  = s2 * (1.0f / DMODEL) - mean * mean;
    float inv  = rsqrtf(var + 1e-5f);
#pragma unroll
    for (int i = 0; i < 4; ++i) {
        int c = threadIdx.x + 128 * i;
        float y = (loc[i] - mean) * inv * lp[c] + lp[DMODEL + c];
        Yf[(size_t)row * DMODEL + c] = y;
        Yb[(size_t)row * DMODEL + c] = f2bf(y);
    }
}

// ---------------- embedding: x@vw + vb + mark@mw + mb ----------------
__global__ void k_embed(const float* __restrict__ xv, const float* __restrict__ xm,
                        const float* __restrict__ vw, const float* __restrict__ vb,
                        const float* __restrict__ mw, const float* __restrict__ mb,
                        float* __restrict__ Yf, bf16_t* __restrict__ Yb, int M, int Kv, int Km) {
    int idx = blockIdx.x * 256 + threadIdx.x;
    if (idx >= M * DMODEL) return;
    int r = idx >> 9, c = idx & 511;
    float s = vb[c] + mb[c];
    for (int k = 0; k < Kv; ++k) s += xv[r * Kv + k] * vw[k * DMODEL + c];
    for (int k = 0; k < Km; ++k) s += xm[r * Km + k] * mw[k * DMODEL + c];
    Yf[idx] = s; Yb[idx] = f2bf(s);
}

// ---------------- weight transpose+convert: W[K,N] f32 -> WT[N,K] bf16 ----------------
__global__ void k_wt(const float* __restrict__ W, bf16_t* __restrict__ WT, int K, int N) {
    int idx = blockIdx.x * 256 + threadIdx.x;
    if (idx >= K * N) return;
    int k = idx / N, n = idx - k * N;
    WT[(size_t)n * K + k] = f2bf(W[idx]);
}
__global__ void k_cvt(const float* __restrict__ W, bf16_t* __restrict__ O, long long n) {
    long long idx = (long long)blockIdx.x * 256 + threadIdx.x;
    if (idx < n) O[idx] = f2bf(W[idx]);
}

// ---------------- head split / merge ----------------
__global__ void k_split_qk(const float* __restrict__ X, bf16_t* __restrict__ O, int L) {
    int idx = blockIdx.x * 256 + threadIdx.x;
    if (idx >= NB * L * DMODEL) return;
    int r = idx >> 9, c = idx & 511;
    int h = c >> 6, d = c & 63, b = r / L, l = r - b * L;
    O[(((size_t)(b * NHEAD + h) * L + l) << 6) + d] = f2bf(X[idx]);
}
__global__ void k_split_vt(const float* __restrict__ X, bf16_t* __restrict__ O, int L) {
    int idx = blockIdx.x * 256 + threadIdx.x;
    if (idx >= NB * L * DMODEL) return;
    int r = idx >> 9, c = idx & 511;
    int h = c >> 6, d = c & 63, b = r / L, l = r - b * L;
    O[((size_t)(b * NHEAD + h) * DK + d) * L + l] = f2bf(X[idx]);
}
__global__ void k_merge(const float* __restrict__ OH, bf16_t* __restrict__ O, int L) {
    int idx = blockIdx.x * 256 + threadIdx.x;
    if (idx >= NB * NHEAD * L * DK) return;
    int d = idx & 63; int rest = idx >> 6;
    int l = rest % L; int bh = rest / L;
    int h = bh & 7, b = bh >> 3;
    O[((size_t)(b * L + l) << 9) + h * DK + d] = f2bf(OH[idx]);
}

// ---------------- conv distill: im2col (circular pad 2), BN stats, BN+ELU+maxpool ----------------
__global__ void k_im2col(const float* __restrict__ X, bf16_t* __restrict__ A, int L) {
    long long idx = (long long)blockIdx.x * 256 + threadIdx.x;
    int Lc = L + 2;
    long long total = (long long)NB * Lc * 1536;
    if (idx >= total) return;
    int col = (int)(idx % 1536);
    long long rt = idx / 1536;
    int t = (int)(rt % Lc); int b = (int)(rt / Lc);
    int i = col / 3, kk = col - i * 3;
    int tl = t + kk - 2; tl = (tl + L) % L;
    A[idx] = f2bf(X[((size_t)(b * L + tl) << 9) + i]);
}
__global__ __launch_bounds__(256) void k_bnstat(const float* __restrict__ Y, float* __restrict__ stat, int Mr) {
    int o = blockIdx.x;
    float s = 0.0f, s2 = 0.0f;
    for (int r = threadIdx.x; r < Mr; r += 256) {
        float v = Y[(size_t)r * DMODEL + o]; s += v; s2 += v * v;
    }
    for (int off = 16; off; off >>= 1) { s += __shfl_xor(s, off, 32); s2 += __shfl_xor(s2, off, 32); }
    __shared__ float sh[16];
    int wid = threadIdx.x >> 5;
    if ((threadIdx.x & 31) == 0) { sh[wid] = s; sh[8 + wid] = s2; }
    __syncthreads();
    if (threadIdx.x == 0) {
        float ts = 0, ts2 = 0;
        for (int w = 0; w < 8; ++w) { ts += sh[w]; ts2 += sh[8 + w]; }
        float mean = ts / Mr;
        stat[o] = mean;
        stat[DMODEL + o] = ts2 / Mr - mean * mean;
    }
}
__global__ void k_pool(const float* __restrict__ Y, const float* __restrict__ stat,
                       const float* __restrict__ g, const float* __restrict__ bb,
                       float* __restrict__ Xf, bf16_t* __restrict__ Xb, int L, int Lp) {
    int idx = blockIdx.x * 256 + threadIdx.x;
    int total = NB * Lp * DMODEL;
    if (idx >= total) return;
    int o = idx & 511; int t2 = idx >> 9;
    int tp = t2 % Lp; int b = t2 / Lp;
    int Lc = L + 2;
    float mean = stat[o];
    float inv = rsqrtf(stat[DMODEL + o] + 1e-5f);
    float gg = g[o], bo = bb[o];
    float best = FLT_NEG;
#pragma unroll
    for (int kk = 0; kk < 3; ++kk) {
        int t = tp * 2 - 1 + kk;
        if (t < 0 || t >= Lc) continue;
        float v = Y[((size_t)(b * Lc + t) << 9) + o];
        v = (v - mean) * inv * gg + bo;
        v = v > 0.0f ? v : (__expf(v) - 1.0f);
        best = fmaxf(best, v);
    }
    Xf[idx] = best; Xb[idx] = f2bf(best);
}

// ==================== host orchestration ====================
extern "C" void kernel_launch(void* const* d_in, const int* in_sizes, int n_in,
                              void* d_out, int out_size, void* d_ws, size_t ws_size,
                              hipStream_t stream) {
    (void)in_sizes; (void)n_in; (void)out_size; (void)ws_size;
    const float* x_enc      = (const float*)d_in[0];
    const float* x_mark_enc = (const float*)d_in[1];
    const float* x_dec      = (const float*)d_in[2];
    const float* x_mark_dec = (const float*)d_in[3];
    const float* enc_vw = (const float*)d_in[4];  const float* enc_vb = (const float*)d_in[5];
    const float* enc_mw = (const float*)d_in[6];  const float* enc_mb = (const float*)d_in[7];
    const float* dec_vw = (const float*)d_in[8];  const float* dec_vb = (const float*)d_in[9];
    const float* dec_mw = (const float*)d_in[10]; const float* dec_mb = (const float*)d_in[11];
    const float* eW  = (const float*)d_in[12]; const float* eb  = (const float*)d_in[13];
    const float* effw1 = (const float*)d_in[14]; const float* effb1 = (const float*)d_in[15];
    const float* effw2 = (const float*)d_in[16]; const float* effb2 = (const float*)d_in[17];
    const float* eln1 = (const float*)d_in[18]; const float* eln2 = (const float*)d_in[19];
    const float* cw = (const float*)d_in[20]; const float* cb = (const float*)d_in[21];
    const float* cbn_g = (const float*)d_in[22]; const float* cbn_b = (const float*)d_in[23];
    const float* enc_norm = (const float*)d_in[24];
    const float* dsW = (const float*)d_in[25]; const float* dsb = (const float*)d_in[26];
    const float* dcW = (const float*)d_in[27]; const float* dcb = (const float*)d_in[28];
    const float* dffw1 = (const float*)d_in[29]; const float* dffb1 = (const float*)d_in[30];
    const float* dffw2 = (const float*)d_in[31]; const float* dffb2 = (const float*)d_in[32];
    const float* dln = (const float*)d_in[33];
    const float* dec_norm = (const float*)d_in[34];
    const float* fcw = (const float*)d_in[35]; const float* fcb = (const float*)d_in[36];

    // ---- workspace bump allocator (each region padded 128KB for tile overshoot) ----
    char* ws = (char*)d_ws;
    size_t off = 0;
    auto alloc = [&](size_t bytes) -> void* {
        off = (off + 255) & ~(size_t)255;
        void* p = ws + off;
        off += bytes + (size_t)131072;
        return p;
    };
    const size_t DD = (size_t)DMODEL * DMODEL;          // 262144
    bf16_t* eWT    = (bf16_t*)alloc(3 * 4 * DD * 2);
    bf16_t* effw1T = (bf16_t*)alloc(3 * (size_t)NDFF * DMODEL * 2);
    bf16_t* effw2T = (bf16_t*)alloc(3 * (size_t)DMODEL * NDFF * 2);
    bf16_t* cwT    = (bf16_t*)alloc(2 * (size_t)DMODEL * 1536 * 2);
    bf16_t* dsWT   = (bf16_t*)alloc(2 * 4 * DD * 2);
    bf16_t* dcWT   = (bf16_t*)alloc(2 * 4 * DD * 2);
    bf16_t* dffw1T = (bf16_t*)alloc(2 * (size_t)NDFF * DMODEL * 2);
    bf16_t* dffw2T = (bf16_t*)alloc(2 * (size_t)DMODEL * NDFF * 2);
    bf16_t* fcwT   = (bf16_t*)alloc(7 * 512 * 2);
    float*  XF   = (float*)alloc((size_t)8192 * 512 * 4);
    bf16_t* XB   = (bf16_t*)alloc((size_t)8192 * 512 * 2);
    float*  TPF  = (float*)alloc((size_t)8192 * 2048 * 4);   // Q/K/V/OH slices or FFN hidden
    bf16_t* TPB  = (bf16_t*)alloc((size_t)8192 * 2048 * 2);  // FFN hidden bf16 / im2col / merged heads
    bf16_t* QH   = (bf16_t*)alloc((size_t)NB * NHEAD * 512 * DK * 2);
    bf16_t* KH   = (bf16_t*)alloc((size_t)NB * NHEAD * 512 * DK * 2);
    bf16_t* VT   = (bf16_t*)alloc((size_t)NB * NHEAD * 512 * DK * 2);
    float*  SF   = (float*)alloc((size_t)16 * 512 * 512 * 4);
    bf16_t* SB   = (bf16_t*)alloc((size_t)16 * 512 * 512 * 2);
    float*  DF   = (float*)alloc((size_t)8192 * 512 * 4);
    float*  MEMF = (float*)alloc((size_t)NB * 130 * 512 * 4);
    bf16_t* MEMB = (bf16_t*)alloc((size_t)NB * 130 * 512 * 2);
    float*  STAT = (float*)alloc(1024 * 4);
    float*  CONVF= (float*)alloc((size_t)NB * 514 * 512 * 4);

    auto gemm = [&](const bf16_t* A, const bf16_t* B, float* C, bf16_t* Cb, const float* bias,
                    int M, int N, int K, int lda, int ldb, int ldc,
                    long long sA, long long sB, long long sC, int batch, float alpha, int relu) {
        dim3 g((N + 127) / 128, (M + 63) / 64, batch);
        k_gemm<<<g, 128, 0, stream>>>(A, B, C, Cb, bias, M, N, K, lda, ldb, ldc, sA, sB, sC, alpha, relu);
    };
    auto wt = [&](const float* W, bf16_t* WT, int K, int N) {
        int n = K * N;
        k_wt<<<(n + 255) / 256, 256, 0, stream>>>(W, WT, K, N);
    };
    auto addln = [&](const float* X, const float* Dl, const float* lp, float* Yf, bf16_t* Yb, int M) {
        k_add_ln<<<M, 128, 0, stream>>>(X, Dl, lp, Yf, Yb, M);
    };

    // ---- weight prep (f32 -> bf16, [K,N] -> [N,K]) ----
    for (int i = 0; i < 3; ++i)
        for (int j = 0; j < 4; ++j)
            wt(eW + ((size_t)i * 4 + j) * DD, eWT + ((size_t)i * 4 + j) * DD, DMODEL, DMODEL);
    for (int i = 0; i < 3; ++i) {
        wt(effw1 + (size_t)i * DMODEL * NDFF, effw1T + (size_t)i * NDFF * DMODEL, DMODEL, NDFF);
        wt(effw2 + (size_t)i * NDFF * DMODEL, effw2T + (size_t)i * DMODEL * NDFF, NDFF, DMODEL);
    }
    { long long n = 2LL * DMODEL * 1536;
      k_cvt<<<(unsigned)((n + 255) / 256), 256, 0, stream>>>(cw, cwT, n); }  // cw already [O, I*3] = [N,K]
    for (int i = 0; i < 2; ++i)
        for (int j = 0; j < 4; ++j) {
            wt(dsW + ((size_t)i * 4 + j) * DD, dsWT + ((size_t)i * 4 + j) * DD, DMODEL, DMODEL);
            wt(dcW + ((size_t)i * 4 + j) * DD, dcWT + ((size_t)i * 4 + j) * DD, DMODEL, DMODEL);
        }
    for (int i = 0; i < 2; ++i) {
        wt(dffw1 + (size_t)i * DMODEL * NDFF, dffw1T + (size_t)i * NDFF * DMODEL, DMODEL, NDFF);
        wt(dffw2 + (size_t)i * NDFF * DMODEL, dffw2T + (size_t)i * DMODEL * NDFF, NDFF, DMODEL);
    }
    wt(fcw, fcwT, DMODEL, 7);

    // ---- attention block (ProbSparse / causal / plain) ----
    auto attention = [&](int Lq, int Lk, const bf16_t* Aq, const bf16_t* Akv,
                         const bf16_t* Wt4, const float* bias4, int mode, float* outDelta) {
        int Mq = NB * Lq, Mkv = NB * Lk;
        float* QF  = TPF;
        float* KF  = TPF + (size_t)8192 * 512;
        float* VF  = TPF + (size_t)8192 * 512 * 2;
        float* OHF = TPF + (size_t)8192 * 512 * 3;
        gemm(Aq,  Wt4 + 0 * DD, QF, 0, bias4 + 0 * DMODEL, Mq,  DMODEL, DMODEL, DMODEL, DMODEL, DMODEL, 0, 0, 0, 1, 1.0f, 0);
        gemm(Akv, Wt4 + 1 * DD, KF, 0, bias4 + 1 * DMODEL, Mkv, DMODEL, DMODEL, DMODEL, DMODEL, DMODEL, 0, 0, 0, 1, 1.0f, 0);
        gemm(Akv, Wt4 + 2 * DD, VF, 0, bias4 + 2 * DMODEL, Mkv, DMODEL, DMODEL, DMODEL, DMODEL, DMODEL, 0, 0, 0, 1, 1.0f, 0);
        int nq = Mq * DMODEL, nk = Mkv * DMODEL;
        k_split_qk<<<(nq + 255) / 256, 256, 0, stream>>>(QF, QH, Lq);
        k_split_qk<<<(nk + 255) / 256, 256, 0, stream>>>(KF, KH, Lk);
        k_split_vt<<<(nk + 255) / 256, 256, 0, stream>>>(VF, VT, Lk);
        for (int bh0 = 0; bh0 < NB * NHEAD; bh0 += 16) {
            gemm(QH + (size_t)bh0 * Lq * DK, KH + (size_t)bh0 * Lk * DK, SF, 0, 0,
                 Lq, Lk, DK, DK, DK, Lk,
                 (long long)Lq * DK, (long long)Lk * DK, (long long)Lq * Lk, 16, 0.125f, 0);
            int rows = 16 * Lq;
            k_softmax<<<(rows + 7) / 8, 256, 0, stream>>>(SF, SB, Lq, Lk, mode, TOPK_K, rows);
            gemm(SB, VT + (size_t)bh0 * DK * Lk, OHF + (size_t)bh0 * Lq * DK, 0, 0,
                 Lq, DK, Lk, Lk, Lk, DK,
                 (long long)Lq * Lk, (long long)DK * Lk, (long long)Lq * DK, 16, 1.0f, 0);
        }
        int no = NB * NHEAD * Lq * DK;
        k_merge<<<(no + 255) / 256, 256, 0, stream>>>(OHF, TPB, Lq);
        gemm(TPB, Wt4 + 3 * DD, outDelta, 0, bias4 + 3 * DMODEL,
             Mq, DMODEL, DMODEL, DMODEL, DMODEL, DMODEL, 0, 0, 0, 1, 1.0f, 0);
    };

    // ================= encoder =================
    {
        int M = NB * 512;
        k_embed<<<(M * DMODEL + 255) / 256, 256, 0, stream>>>(
            x_enc, x_mark_enc, enc_vw, enc_vb, enc_mw, enc_mb, XF, XB, M, 7, 4);
    }
    int L = 512;
    for (int i = 0; i < 3; ++i) {
        int M = NB * L;
        attention(L, L, XB, XB, eWT + (size_t)i * 4 * DD, eb + (size_t)i * 4 * DMODEL, 1, DF);
        addln(XF, DF, eln1 + (size_t)i * 2 * DMODEL, XF, XB, M);
        gemm(XB, effw1T + (size_t)i * NDFF * DMODEL, TPF, TPB, effb1 + (size_t)i * NDFF,
             M, NDFF, DMODEL, DMODEL, DMODEL, NDFF, 0, 0, 0, 1, 1.0f, 1);
        gemm(TPB, effw2T + (size_t)i * DMODEL * NDFF, DF, 0, effb2 + (size_t)i * DMODEL,
             M, DMODEL, NDFF, NDFF, NDFF, DMODEL, 0, 0, 0, 1, 1.0f, 0);
        addln(XF, DF, eln2 + (size_t)i * 2 * DMODEL, XF, XB, M);
        if (i < 2) {
            int Lc = L + 2;
            int Lp = (L + 1) / 2 + 1;       // 512 -> 257, 257 -> 130
            long long ni = (long long)NB * Lc * 1536;
            k_im2col<<<(unsigned)((ni + 255) / 256), 256, 0, stream>>>(XF, TPB, L);
            gemm(TPB, cwT + (size_t)i * DMODEL * 1536, CONVF, 0, cb + (size_t)i * DMODEL,
                 NB * Lc, DMODEL, 1536, 1536, 1536, DMODEL, 0, 0, 0, 1, 1.0f, 0);
            k_bnstat<<<DMODEL, 256, 0, stream>>>(CONVF, STAT, NB * Lc);
            int np = NB * Lp * DMODEL;
            k_pool<<<(np + 255) / 256, 256, 0, stream>>>(
                CONVF, STAT, cbn_g + (size_t)i * DMODEL, cbn_b + (size_t)i * DMODEL, XF, XB, L, Lp);
            L = Lp;
        }
    }
    int Lmem = L;                            // 130
    addln(XF, 0, enc_norm, MEMF, MEMB, NB * Lmem);

    // ================= decoder =================
    {
        int M = NB * 256;
        k_embed<<<(M * DMODEL + 255) / 256, 256, 0, stream>>>(
            x_dec, x_mark_dec, dec_vw, dec_vb, dec_mw, dec_mb, XF, XB, M, 7, 4);
    }
    for (int i = 0; i < 2; ++i) {
        int M = NB * 256;
        attention(256, 256, XB, XB, dsWT + (size_t)i * 4 * DD, dsb + (size_t)i * 4 * DMODEL, 2, DF);
        addln(XF, DF, dln + (size_t)(i * 3 + 0) * 2 * DMODEL, XF, XB, M);
        attention(256, Lmem, XB, MEMB, dcWT + (size_t)i * 4 * DD, dcb + (size_t)i * 4 * DMODEL, 0, DF);
        addln(XF, DF, dln + (size_t)(i * 3 + 1) * 2 * DMODEL, XF, XB, M);
        gemm(XB, dffw1T + (size_t)i * NDFF * DMODEL, TPF, TPB, dffb1 + (size_t)i * NDFF,
             M, NDFF, DMODEL, DMODEL, DMODEL, NDFF, 0, 0, 0, 1, 1.0f, 1);
        gemm(TPB, dffw2T + (size_t)i * DMODEL * NDFF, DF, 0, dffb2 + (size_t)i * DMODEL,
             M, DMODEL, NDFF, NDFF, NDFF, DMODEL, 0, 0, 0, 1, 1.0f, 0);
        addln(XF, DF, dln + (size_t)(i * 3 + 2) * 2 * DMODEL, XF, XB, M);
    }
    addln(XF, 0, dec_norm, XF, XB, NB * 256);
    gemm(XB, fcwT, (float*)d_out, 0, fcb, NB * 256, 7, DMODEL, DMODEL, DMODEL, 7, 0, 0, 0, 1, 1.0f, 0);
}
